// MLPPredictor4deepwalk_22832046146011
// MI455X (gfx1250) — compile-verified, matrix operand order checked by reading, not run on hardware
//
#include <hip/hip_runtime.h>
#include <hip/hip_bf16.h>

typedef __attribute__((ext_vector_type(16))) __bf16 v16bf;
typedef __attribute__((ext_vector_type(8)))  float  v8f;
typedef __attribute__((ext_vector_type(4)))  unsigned int v4u;

#define N_NODES 50000
#define H_FEATS 128
#define N_EDGES 600000

// W1 packed fragments: 64 fragments (nt 0..7, kt 0..7), each 32 lanes x 16 bf16 = 1KB
#define W1P_HALFS (64 * 32 * 16)   // 32768 halves = 64KB

static __device__ __forceinline__ unsigned short f32_to_bf16_rne(float f) {
    unsigned int u = __float_as_uint(f);
    unsigned int r = 0x7FFFu + ((u >> 16) & 1u);
    return (unsigned short)((u + r) >> 16);
}

// --- prep: h (f32) -> bf16, two elements per thread packed as one dword ---
__global__ void cvt_h_kernel(const float* __restrict__ h,
                             unsigned int* __restrict__ hb_u32, int n_pairs) {
    int i = blockIdx.x * blockDim.x + threadIdx.x;
    if (i < n_pairs) {
        unsigned int lo = f32_to_bf16_rne(h[2 * i]);
        unsigned int hi = f32_to_bf16_rne(h[2 * i + 1]);
        hb_u32[i] = lo | (hi << 16);
    }
}

// --- prep: W1 (f32 [128 out, 256 in]) -> bf16 WMMA B-fragment layout ---
// Fragment f = nt*8+kt holds B[k=kt*32..+32, n=nt*16..+16].
// 16-bit B layout (mirror of A, lane = n): lanes 0-15 hold K {0..7,16..23},
// lanes 16-31 hold K {8..15,24..31}; half j -> K = base8 + (j&7) + (j>>3)*16.
__global__ void pack_w1_kernel(const float* __restrict__ W1,
                               unsigned short* __restrict__ w1p) {
    int p = blockIdx.x * blockDim.x + threadIdx.x;
    if (p >= W1P_HALFS) return;
    int j    = p & 15;
    int lane = (p >> 4) & 31;
    int frag = p >> 9;            // 0..63
    int nt   = frag >> 3;
    int kt   = frag & 7;
    int n    = nt * 16 + (lane & 15);
    int k    = kt * 32 + ((lane < 16) ? 0 : 8) + (j & 7) + ((j >> 3) << 4);
    w1p[p] = f32_to_bf16_rne(W1[n * 256 + k]);
}

// --- main: 8 waves/block, 4 tiles of 16 edges per wave ---
#define TILES_PER_BLOCK 32

__global__ __launch_bounds__(256)
void edge_mlp_kernel(const unsigned short* __restrict__ hb,
                     const unsigned short* __restrict__ w1p,
                     const float* __restrict__ W1_b,
                     const float* __restrict__ W2_w,
                     const float* __restrict__ W2_b,
                     const int*   __restrict__ esrc,
                     const int*   __restrict__ edst,
                     float*       __restrict__ out) {
    __shared__ __align__(32) unsigned short s_w1[W1P_HALFS];  // 64KB

    // Stage packed W1 into LDS. Explicit ds stores (NOT inline-asm async):
    // the optimizer must see the writes, otherwise the B-fragment ds_loads
    // are folded to undef and the whole WMMA grid collapses (rounds 2-3 bug).
    {
        const v4u* g = (const v4u*)w1p;
        v4u*       s = (v4u*)s_w1;
        for (int i = threadIdx.x; i < W1P_HALFS / 8; i += blockDim.x) s[i] = g[i];
    }
    __syncthreads();

    const int lane    = threadIdx.x & 31;
    const int wave    = threadIdx.x >> 5;
    const int m       = lane & 15;     // edge row within tile
    const int halfsel = lane >> 4;     // K-range selector (A layout)

    float w2v[8], b1v[8];
#pragma unroll
    for (int nt = 0; nt < 8; ++nt) {
        w2v[nt] = W2_w[nt * 16 + m];
        b1v[nt] = W1_b[nt * 16 + m];
    }
    const float b2 = W2_b[0];

    for (int t = 0; t < 4; ++t) {
        int tile = blockIdx.x * TILES_PER_BLOCK + wave * 4 + t;
        int e0   = tile * 16;
        if (e0 >= N_EDGES) break;                 // wave-uniform

        int e = e0 + m;
        if (e >= N_EDGES) e = N_EDGES - 1;        // clamp loads; stores guarded
        const int nsrc = esrc[e];
        const int ndst = edst[e];

        // 8 named accumulators -> 8 independent WMMA streams.
        v8f acc0 = {}, acc1 = {}, acc2 = {}, acc3 = {};
        v8f acc4 = {}, acc5 = {}, acc6 = {}, acc7 = {};

#pragma unroll
        for (int kt = 0; kt < 8; ++kt) {
            // K 0..127 of he comes from h[src], 128..255 from h[dst]
            const int node  = (kt < 4) ? nsrc : ndst;
            const int kbase = (kt & 3) * 32 + halfsel * 8;
            const unsigned short* ap = hb + (size_t)node * H_FEATS + kbase;

            union { v16bf v; v4u q[2]; } A;
            A.q[0] = *(const v4u*)(ap);          // K kbase+0..7
            A.q[1] = *(const v4u*)(ap + 16);     // K kbase+16..23

            // Burst-load 8 B fragments (16 ds_load_b128), one wait, then 8
            // independent back-to-back WMMAs (no per-WMMA LDS stall).
#define BLOAD(nt) v16bf B##nt = *(const v16bf*)(s_w1 + (((nt) * 8 + kt) * 32 + lane) * 16)
            BLOAD(0); BLOAD(1); BLOAD(2); BLOAD(3);
            BLOAD(4); BLOAD(5); BLOAD(6); BLOAD(7);
#undef BLOAD
#if __has_builtin(__builtin_amdgcn_sched_barrier)
            __builtin_amdgcn_sched_barrier(0);   // keep loads above the WMMA burst
#endif
#define WM(nt) acc##nt = __builtin_amdgcn_wmma_f32_16x16x32_bf16( \
                false, A.v, false, B##nt, (short)0, acc##nt, false, false)
            WM(0); WM(1); WM(2); WM(3);
            WM(4); WM(5); WM(6); WM(7);
#undef WM
        }

        // layer 2: bias + relu + dot with W2, reduce across the 16-lane half
        v8f accarr[8] = { acc0, acc1, acc2, acc3, acc4, acc5, acc6, acc7 };
        float p[8];
#pragma unroll
        for (int r = 0; r < 8; ++r) {
            float s = 0.0f;
#pragma unroll
            for (int nt = 0; nt < 8; ++nt) {
                float hval = accarr[nt][r] + b1v[nt];
                hval = hval > 0.0f ? hval : 0.0f;
                s += hval * w2v[nt];
            }
            p[r] = s;
        }
#pragma unroll
        for (int mask = 1; mask <= 8; mask <<= 1)
#pragma unroll
            for (int r = 0; r < 8; ++r) p[r] += __shfl_xor(p[r], mask, 32);

        if (m == 0) {
            int base = e0 + halfsel * 8;          // lane0 -> rows 0..7, lane16 -> 8..15
#pragma unroll
            for (int r = 0; r < 8; ++r) {
                int eo = base + r;
                if (eo < N_EDGES) out[eo] = p[r] + b2;
            }
        }
    }
}

extern "C" void kernel_launch(void* const* d_in, const int* in_sizes, int n_in,
                              void* d_out, int out_size, void* d_ws, size_t ws_size,
                              hipStream_t stream) {
    const float* h    = (const float*)d_in[0];
    const float* W1_w = (const float*)d_in[1];
    const float* W1_b = (const float*)d_in[2];
    const float* W2_w = (const float*)d_in[3];
    const float* W2_b = (const float*)d_in[4];
    const int*   esrc = (const int*)d_in[5];
    const int*   edst = (const int*)d_in[6];
    float* out = (float*)d_out;

    unsigned short* w1p = (unsigned short*)d_ws;                   // 64KB
    unsigned short* hb  = (unsigned short*)((char*)d_ws + 65536);  // 12.8MB bf16 h

    // prep: pack W1 and convert h to bf16
    pack_w1_kernel<<<(W1P_HALFS + 255) / 256, 256, 0, stream>>>(W1_w, w1p);
    int n_pairs = (N_NODES * H_FEATS) / 2;
    cvt_h_kernel<<<(n_pairs + 255) / 256, 256, 0, stream>>>(h, (unsigned int*)hb, n_pairs);

    // main: 16-edge tiles, 32 tiles per block
    int n_tiles = (N_EDGES + 15) / 16;
    int blocks  = (n_tiles + TILES_PER_BLOCK - 1) / TILES_PER_BLOCK;
    edge_mlp_kernel<<<blocks, 256, 0, stream>>>(hb, w1p, W1_b, W2_w, W2_b,
                                                esrc, edst, out);
}
